// SelfAttention_22445499089373
// MI455X (gfx1250) — matrix operand order repaired and checked by reading, not compiled, not run on hardware
//
#include <hip/hip_runtime.h>
#include <stdint.h>

// ---------------------------------------------------------------------------
// Types for CDNA5 WMMA (wave32): 16x16x32 bf16 -> f32
// ---------------------------------------------------------------------------
typedef __attribute__((ext_vector_type(16))) __bf16 v16bf;
typedef __attribute__((ext_vector_type(8)))  float  v8f;

struct __align__(16) U4 { uint32_t x, y, z, w; };

#define SEQ   1024
#define DIMK  1024
#define NBATCH 8

// ---------------------------------------------------------------------------
// Helpers
// ---------------------------------------------------------------------------
__device__ __forceinline__ float fqv(float x, float amax) {
    // fake_quant with bit=8: qmax=127, clip [-128,127], round-nearest-even
    float s = 127.0f / fmaxf(amax, 1e-8f);
    float q = rintf(x * s);
    q = fminf(fmaxf(q, -128.0f), 127.0f);
    return q / s;
}

__device__ __forceinline__ uint16_t f2bf(float f) {
    uint32_t u = __float_as_uint(f);
    uint32_t lsb = (u >> 16) & 1u;
    u += 0x7fffu + lsb;           // round-to-nearest-even
    return (uint16_t)(u >> 16);
}

__device__ __forceinline__ void atomicMaxPosF(float* addr, float v) {
    // v >= 0 so the uint bit pattern is order-preserving
    atomicMax((unsigned int*)addr, __float_as_uint(v));
}

// CDNA5 async global -> LDS copy, 16B per lane, GV addressing mode
__device__ __forceinline__ void async_copy16(uint32_t ldsOff, const void* gaddr) {
    asm volatile("global_load_async_to_lds_b128 %0, %1, off"
                 :: "v"(ldsOff), "v"(gaddr)
                 : "memory");
}
__device__ __forceinline__ void wait_async_le4() {
    asm volatile("s_wait_asynccnt 0x4" ::: "memory");
}
__device__ __forceinline__ void wait_async_0() {
    asm volatile("s_wait_asynccnt 0x0" ::: "memory");
}

// ---------------------------------------------------------------------------
// Small elementwise / reduction kernels
// ---------------------------------------------------------------------------
__global__ void init_scal_k(float* scal) {
    if (threadIdx.x < 8) scal[threadIdx.x] = 0.0f;
}

__global__ void absmax_k(const float* __restrict__ x, int n, float* out) {
    __shared__ float red[256];
    int tid = threadIdx.x;
    float m = 0.f;
    for (int i = blockIdx.x * blockDim.x + tid; i < n; i += gridDim.x * blockDim.x)
        m = fmaxf(m, fabsf(x[i]));
    red[tid] = m; __syncthreads();
    for (int s = 128; s > 0; s >>= 1) {
        if (tid < s) red[tid] = fmaxf(red[tid], red[tid + s]);
        __syncthreads();
    }
    if (tid == 0) atomicMaxPosF(out, red[0]);
}

__global__ void quantw_k(const float* __restrict__ w, const float* __restrict__ amax,
                         uint16_t* __restrict__ o, int n) {
    int i = blockIdx.x * blockDim.x + threadIdx.x;
    if (i < n) o[i] = f2bf(fqv(w[i], *amax));
}

__global__ void f2bf_k(const float* __restrict__ x, uint16_t* __restrict__ o, int n) {
    int i = blockIdx.x * blockDim.x + threadIdx.x;
    if (i < n) o[i] = f2bf(x[i]);
}

__global__ void quantp_k(const float* __restrict__ p, const float* __restrict__ amax,
                         uint16_t* __restrict__ o, int n) {
    int i = blockIdx.x * blockDim.x + threadIdx.x;
    if (i < n) o[i] = f2bf(fqv(p[i], *amax));
}

__global__ void finalfq_k(float* __restrict__ o, const float* __restrict__ amax, int n) {
    int i = blockIdx.x * blockDim.x + threadIdx.x;
    if (i < n) o[i] = fqv(o[i], *amax);
}

// ---------------------------------------------------------------------------
// Tiled NT GEMM:  C[m,n] = sum_k A[m,k] * B[n,k]   (both K-contiguous, bf16)
// 64x64 tile per 128-thread block (4 waves, 2x2), K step 32,
// double-buffered LDS filled with global_load_async_to_lds_b128,
// compute via v_wmma_f32_16x16x32_bf16 (4 WMMA tiles per wave).
// Output: f32 (+optional |.| amax atomic)  OR  bf16 (optionally transposed
// per-batch for building V^T).
// ---------------------------------------------------------------------------
__global__ void __launch_bounds__(128)
gemm_nt_k(const uint16_t* __restrict__ A, long long sA,
          const uint16_t* __restrict__ B, long long sB,
          float* __restrict__ Cf, uint16_t* __restrict__ Cb, long long sC,
          int K, int lda, int ldb, int ldc,
          int transOut, int tBatchM, int Ntot,
          float* __restrict__ amax)
{
    __shared__ __align__(16) uint16_t Asub[2][64][32];
    __shared__ __align__(16) uint16_t Bsub[2][64][32];

    const int tid  = threadIdx.x;
    const int lane = tid & 31;
    const int wid  = tid >> 5;            // 0..3
    const int wm   = (wid >> 1) * 32;     // wave row offset within 64x64 tile
    const int wn   = (wid & 1) * 32;      // wave col offset
    const int l16  = lane & 15;
    const int g    = lane >> 4;           // lane group 0/1

    const long long bz = blockIdx.z;
    const uint16_t* Ab = A + bz * sA;
    const uint16_t* Bb = B + bz * sB;
    const int blockM = blockIdx.y * 64;
    const int blockN = blockIdx.x * 64;

    // issue 4 async b128 per thread (2 for A tile, 2 for B tile)
    auto prefetch = [&](int kt, int buf) {
        const int k0 = kt * 32;
        #pragma unroll
        for (int h = 0; h < 2; ++h) {
            const int c   = tid + h * 128;       // 0..255 chunk id
            const int row = c >> 2;              // 0..63
            const int cc  = c & 3;               // 0..3 (8 bf16 each)
            const uint16_t* ga = Ab + (long long)(blockM + row) * lda + k0 + cc * 8;
            async_copy16((uint32_t)(uintptr_t)&Asub[buf][row][cc * 8], ga);
            const uint16_t* gb = Bb + (long long)(blockN + row) * ldb + k0 + cc * 8;
            async_copy16((uint32_t)(uintptr_t)&Bsub[buf][row][cc * 8], gb);
        }
    };

    v8f acc[2][2];
    #pragma unroll
    for (int tm = 0; tm < 2; ++tm)
        #pragma unroll
        for (int tn = 0; tn < 2; ++tn)
            #pragma unroll
            for (int r = 0; r < 8; ++r) acc[tm][tn][r] = 0.0f;

    const int KT = K >> 5;
    prefetch(0, 0);

    for (int kt = 0; kt < KT; ++kt) {
        const int cur = kt & 1;
        const bool more = (kt + 1 < KT);
        if (more) prefetch(kt + 1, cur ^ 1);
        if (more) wait_async_le4(); else wait_async_0();
        __syncthreads();

        // A fragment (16-bit A 16x32 layout): lane l16 = row, group g selects
        // K ranges [8g,8g+8) and [16+8g,16+8g+8)
        union AF { v16bf v; U4 u[2]; } afr[2], bfr[2];
        #pragma unroll
        for (int tm = 0; tm < 2; ++tm) {
            const int row = wm + tm * 16 + l16;
            afr[tm].u[0] = *(const U4*)&Asub[cur][row][g * 8];
            afr[tm].u[1] = *(const U4*)&Asub[cur][row][16 + g * 8];
        }
        // B fragment (16-bit B 32x16 layout): lane l16 = column, group g
        // selects K ranges [8g,8g+8) and [16+8g,16+8g+8)
        #pragma unroll
        for (int tn = 0; tn < 2; ++tn) {
            const int col = wn + tn * 16 + l16;
            bfr[tn].u[0] = *(const U4*)&Bsub[cur][col][g * 8];
            bfr[tn].u[1] = *(const U4*)&Bsub[cur][col][16 + g * 8];
        }

        #pragma unroll
        for (int tm = 0; tm < 2; ++tm)
            #pragma unroll
            for (int tn = 0; tn < 2; ++tn)
                acc[tm][tn] = __builtin_amdgcn_wmma_f32_16x16x32_bf16(
                    false, afr[tm].v, false, bfr[tn].v,
                    (short)0, acc[tm][tn], false, false);

        __syncthreads();
    }

    // C layout: VGPR r holds M = r (lanes 0-15) / r+8 (lanes 16-31), N = lane%16
    float localMax = 0.0f;
    #pragma unroll
    for (int tm = 0; tm < 2; ++tm) {
        #pragma unroll
        for (int tn = 0; tn < 2; ++tn) {
            const int rowBase = blockM + wm + tm * 16 + g * 8;
            const int col     = blockN + wn + tn * 16 + l16;
            #pragma unroll
            for (int r = 0; r < 8; ++r) {
                const int row = rowBase + r;
                const float v = acc[tm][tn][r];
                if (Cf) {
                    Cf[bz * sC + (long long)row * ldc + col] = v;
                    localMax = fmaxf(localMax, fabsf(v));
                } else if (transOut) {
                    const int bb = row / tBatchM;
                    const int sl = row - bb * tBatchM;
                    Cb[(long long)bb * Ntot * tBatchM + (long long)col * tBatchM + sl] = f2bf(v);
                } else {
                    Cb[bz * sC + (long long)row * ldc + col] = f2bf(v);
                }
            }
        }
    }
    if (Cf && amax) atomicMaxPosF(amax, localMax);
}

// ---------------------------------------------------------------------------
// Fused fake-quant chain + row softmax, in place on S rows of length 1024.
//   w = fq(fq(S,a1)*nq, a1*nq); P = softmax_row(w); track amax(P).
// (fq preserves the tensor max, so amax(fq(S)*nq) == amax(S)*nq analytically.)
// ---------------------------------------------------------------------------
__global__ void __launch_bounds__(256)
softmax_fq_k(float* __restrict__ S, const float* __restrict__ amaxS,
             float nq, float* __restrict__ amaxP)
{
    __shared__ float red[256];
    const int row = blockIdx.x;
    const int tid = threadIdx.x;
    float* r = S + (long long)row * SEQ;

    const float a1 = fmaxf(*amaxS, 1e-8f);
    const float a2 = fmaxf(a1 * nq, 1e-8f);

    float w[4];
    float m = -3.4e38f;
    #pragma unroll
    for (int i = 0; i < 4; ++i) {
        const int j = tid + i * 256;
        const float t = fqv(r[j], a1);
        w[i] = fqv(t * nq, a2);
        m = fmaxf(m, w[i]);
    }
    red[tid] = m; __syncthreads();
    for (int s = 128; s > 0; s >>= 1) {
        if (tid < s) red[tid] = fmaxf(red[tid], red[tid + s]);
        __syncthreads();
    }
    const float rowmax = red[0]; __syncthreads();

    float e[4]; float sum = 0.f;
    #pragma unroll
    for (int i = 0; i < 4; ++i) { e[i] = __expf(w[i] - rowmax); sum += e[i]; }
    red[tid] = sum; __syncthreads();
    for (int s = 128; s > 0; s >>= 1) {
        if (tid < s) red[tid] += red[tid + s];
        __syncthreads();
    }
    const float inv = 1.0f / red[0]; __syncthreads();

    float pmax = 0.f;
    #pragma unroll
    for (int i = 0; i < 4; ++i) {
        const float p = e[i] * inv;
        r[tid + i * 256] = p;
        pmax = fmaxf(pmax, p);
    }
    red[tid] = pmax; __syncthreads();
    for (int s = 128; s > 0; s >>= 1) {
        if (tid < s) red[tid] = fmaxf(red[tid], red[tid + s]);
        __syncthreads();
    }
    if (tid == 0) atomicMaxPosF(amaxP, red[0]);
}

// ---------------------------------------------------------------------------
// Host-side orchestration
// ---------------------------------------------------------------------------
extern "C" void kernel_launch(void* const* d_in, const int* in_sizes, int n_in,
                              void* d_out, int out_size, void* d_ws, size_t ws_size,
                              hipStream_t stream)
{
    (void)in_sizes; (void)n_in; (void)out_size; (void)ws_size;

    const float* x  = (const float*)d_in[0];
    const float* Wq = (const float*)d_in[1];
    const float* Wk = (const float*)d_in[2];
    const float* Wv = (const float*)d_in[3];
    float* out = (float*)d_out;

    const int BS   = NBATCH * SEQ;          // 8192 rows
    const int nW   = DIMK * DIMK;           // 1,048,576
    const int nX   = BS * DIMK;             // 8,388,608
    const long long batchStride = (long long)SEQ * DIMK;  // 1M elems
    const float nq = 0.03125f;              // fake_quant(1/sqrt(1024)) == 1/32

    // workspace carve-up
    char* w = (char*)d_ws;
    float*    scal = (float*)w;                         size_t off = 256;
    uint16_t* Xb   = (uint16_t*)(w + off);              off += (size_t)nX * 2;
    uint16_t* Wqb  = (uint16_t*)(w + off);              off += (size_t)nW * 2;
    uint16_t* Wkb  = (uint16_t*)(w + off);              off += (size_t)nW * 2;
    uint16_t* Wvb  = (uint16_t*)(w + off);              off += (size_t)nW * 2;
    uint16_t* Qb   = (uint16_t*)(w + off);              off += (size_t)nX * 2;
    uint16_t* Kb   = (uint16_t*)(w + off);              off += (size_t)nX * 2;
    uint16_t* Vtb  = (uint16_t*)(w + off);              off += (size_t)nX * 2;
    float*    Sws  = (float*)(w + off);                 off += (size_t)nX * 4;
    uint16_t* Pq   = (uint16_t*)(w + off);              off += (size_t)nX * 2;

    float* amaxWq = scal + 0;
    float* amaxWk = scal + 1;
    float* amaxWv = scal + 2;
    float* amaxS  = scal + 3;
    float* amaxP  = scal + 4;
    float* amaxO  = scal + 5;

    // 0) zero amax accumulators
    init_scal_k<<<1, 32, 0, stream>>>(scal);

    // 1) per-tensor abs-max of the weights
    absmax_k<<<1024, 256, 0, stream>>>(Wq, nW, amaxWq);
    absmax_k<<<1024, 256, 0, stream>>>(Wk, nW, amaxWk);
    absmax_k<<<1024, 256, 0, stream>>>(Wv, nW, amaxWv);

    // 2) fake-quantize weights -> bf16; activations -> bf16
    quantw_k<<<(nW + 255) / 256, 256, 0, stream>>>(Wq, amaxWq, Wqb, nW);
    quantw_k<<<(nW + 255) / 256, 256, 0, stream>>>(Wk, amaxWk, Wkb, nW);
    quantw_k<<<(nW + 255) / 256, 256, 0, stream>>>(Wv, amaxWv, Wvb, nW);
    f2bf_k  <<<(nX + 255) / 256, 256, 0, stream>>>(x, Xb, nX);

    // 3) projections: Q = X * Wq^T, K = X * Wk^T  (bf16 out, row-major)
    dim3 gProj(DIMK / 64, BS / 64, 1);
    gemm_nt_k<<<gProj, 128, 0, stream>>>(Xb, 0, Wqb, 0, nullptr, Qb, 0,
                                         DIMK, DIMK, DIMK, DIMK,
                                         0, SEQ, DIMK, nullptr);
    gemm_nt_k<<<gProj, 128, 0, stream>>>(Xb, 0, Wkb, 0, nullptr, Kb, 0,
                                         DIMK, DIMK, DIMK, DIMK,
                                         0, SEQ, DIMK, nullptr);
    // V projection stored transposed per batch: Vt[b][v][s]
    gemm_nt_k<<<gProj, 128, 0, stream>>>(Xb, 0, Wvb, 0, nullptr, Vtb, 0,
                                         DIMK, DIMK, DIMK, DIMK,
                                         1, SEQ, DIMK, nullptr);

    // 4) scores: S[b] = Q[b] * K[b]^T  (f32 out + amax)
    dim3 gAttn(SEQ / 64, SEQ / 64, NBATCH);
    gemm_nt_k<<<gAttn, 128, 0, stream>>>(Qb, batchStride, Kb, batchStride,
                                         Sws, nullptr, batchStride,
                                         DIMK, DIMK, DIMK, SEQ,
                                         0, SEQ, SEQ, amaxS);

    // 5) fq chain + softmax (in place on Sws), track amax(P)
    softmax_fq_k<<<NBATCH * SEQ, 256, 0, stream>>>(Sws, amaxS, nq, amaxP);

    // 6) fake-quantize probabilities -> bf16
    quantp_k<<<(nX + 255) / 256, 256, 0, stream>>>(Sws, amaxP, Pq, nX);

    // 7) output: O[b] = P[b] * V[b]  == P * (Vt)^T  (f32 into d_out + amax)
    gemm_nt_k<<<gAttn, 128, 0, stream>>>(Pq, batchStride, Vtb, batchStride,
                                         out, nullptr, batchStride,
                                         SEQ, SEQ, SEQ, DIMK,
                                         0, SEQ, DIMK, amaxO);

    // 8) final fake-quant in place on d_out
    finalfq_k<<<(nX + 255) / 256, 256, 0, stream>>>(out, amaxO, nX);
}
